// GNBlock_8461085573692
// MI455X (gfx1250) — matrix operand order repaired and checked by reading, not compiled, not run on hardware
//
#include <hip/hip_runtime.h>
#include <hip/hip_bf16.h>

#define Nn 50000
#define Ee 400000
#define Bb 8

typedef __bf16 bf16t;
typedef __attribute__((ext_vector_type(16))) __bf16 v16bf;
typedef __attribute__((ext_vector_type(8)))  __bf16 v8bf;
typedef __attribute__((ext_vector_type(8)))  float  v8f;

// ---------------- helpers ----------------

__device__ __forceinline__ void atomAddG(float* p, float v) {
    __hip_atomic_fetch_add(p, v, __ATOMIC_RELAXED, __HIP_MEMORY_SCOPE_AGENT);
}
__device__ __forceinline__ void atomAddL(float* p, float v) {
    __hip_atomic_fetch_add(p, v, __ATOMIC_RELAXED, __HIP_MEMORY_SCOPE_WORKGROUP);
}

// Build a 16x32 bf16 A-fragment from a per-lane fp32 row segment (32 floats).
// ISA layout: lanes 0-15 = rows, K 0..7 + 16..23; lanes 16-31 = rows, K 8..15 + 24..31.
__device__ __forceinline__ v16bf a_from_f32(const float* seg, int lane) {
    int off = ((lane >> 4) << 3);
    const float4* p = (const float4*)(seg + off);
    float4 u0 = p[0], u1 = p[1], u2 = p[4], u3 = p[5];
    v16bf a;
    a[0]  = (bf16t)u0.x; a[1]  = (bf16t)u0.y; a[2]  = (bf16t)u0.z; a[3]  = (bf16t)u0.w;
    a[4]  = (bf16t)u1.x; a[5]  = (bf16t)u1.y; a[6]  = (bf16t)u1.z; a[7]  = (bf16t)u1.w;
    a[8]  = (bf16t)u2.x; a[9]  = (bf16t)u2.y; a[10] = (bf16t)u2.z; a[11] = (bf16t)u2.w;
    a[12] = (bf16t)u3.x; a[13] = (bf16t)u3.y; a[14] = (bf16t)u3.z; a[15] = (bf16t)u3.w;
    return a;
}

// A fragment from a bf16 LDS tile stored row-major [rows][128].
__device__ __forceinline__ v16bf a_from_lds(const bf16t* h, int r, int kb) {
    const bf16t* pr = h + r * 128 + kb;
    v8bf lo = *(const v8bf*)(pr);
    v8bf hh = *(const v8bf*)(pr + 16);
    v16bf a;
#pragma unroll
    for (int i = 0; i < 8; ++i) { a[i] = lo[i]; a[i + 8] = hh[i]; }
    return a;
}

// B fragment: weights stored transposed+bf16, Wt[n][k]. Lanes 0-15: K 0..7 + 16..23 of the
// 32-chunk, lanes 16-31: K 8..15 + 24..31; n = nt*16 + (lane&15). Two 8-bf16 (16B) loads.
__device__ __forceinline__ v16bf b_frag(const bf16t* Wt, int Kdim, int nt, int kk, int lane) {
    const bf16t* pb = Wt + (size_t)(nt * 16 + (lane & 15)) * Kdim + kk * 32 + ((lane >> 4) << 3);
    v8bf lo = *(const v8bf*)(pb);
    v8bf hh = *(const v8bf*)(pb + 16);
    v16bf b;
#pragma unroll
    for (int i = 0; i < 8; ++i) { b[i] = lo[i]; b[i + 8] = hh[i]; }
    return b;
}

#define WMMA_BF16(A, B, C) __builtin_amdgcn_wmma_f32_16x16x32_bf16(false, A, false, B, (short)0, C, false, false)

// One K-step for two 16-row tiles: preload 4 B fragments (clause of 8 b128 loads,
// one wait), then 8 WMMAs back-to-back; repeat for the second nt half.
__device__ __forceinline__ void wmma_step2(v8f c0[8], v8f c1[8], v16bf a0, v16bf a1,
                                           const bf16t* Wt, int Kdim, int kk, int lane) {
#pragma unroll
    for (int h2 = 0; h2 < 2; ++h2) {
        v16bf b0 = b_frag(Wt, Kdim, h2 * 4 + 0, kk, lane);
        v16bf b1 = b_frag(Wt, Kdim, h2 * 4 + 1, kk, lane);
        v16bf b2 = b_frag(Wt, Kdim, h2 * 4 + 2, kk, lane);
        v16bf b3 = b_frag(Wt, Kdim, h2 * 4 + 3, kk, lane);
        c0[h2 * 4 + 0] = WMMA_BF16(a0, b0, c0[h2 * 4 + 0]);
        c1[h2 * 4 + 0] = WMMA_BF16(a1, b0, c1[h2 * 4 + 0]);
        c0[h2 * 4 + 1] = WMMA_BF16(a0, b1, c0[h2 * 4 + 1]);
        c1[h2 * 4 + 1] = WMMA_BF16(a1, b1, c1[h2 * 4 + 1]);
        c0[h2 * 4 + 2] = WMMA_BF16(a0, b2, c0[h2 * 4 + 2]);
        c1[h2 * 4 + 2] = WMMA_BF16(a1, b2, c1[h2 * 4 + 2]);
        c0[h2 * 4 + 3] = WMMA_BF16(a0, b3, c0[h2 * 4 + 3]);
        c1[h2 * 4 + 3] = WMMA_BF16(a1, b3, c1[h2 * 4 + 3]);
    }
}

// bias+(relu)+round-trip through wave-private 32x128 LDS tile, then next 128x128 layer.
__device__ __forceinline__ void next_layer2(v8f c0[8], v8f c1[8], const float* bias, bool relu,
                                            const bf16t* Wt, bf16t* h, int lane) {
    int nlo = lane & 15, hi = lane >> 4;
    v8f zero = {};
#pragma unroll
    for (int nt = 0; nt < 8; ++nt) {
        int n = nt * 16 + nlo;
        float b = bias[n];
#pragma unroll
        for (int g = 0; g < 8; ++g) {
            int M = g + hi * 8;             // C/D layout: lane half selects M block
            float v0 = c0[nt][g] + b;
            float v1 = c1[nt][g] + b;
            if (relu) { v0 = fmaxf(v0, 0.0f); v1 = fmaxf(v1, 0.0f); }
            h[M * 128 + n]        = (bf16t)v0;
            h[(16 + M) * 128 + n] = (bf16t)v1;
        }
    }
#pragma unroll
    for (int nt = 0; nt < 8; ++nt) { c0[nt] = zero; c1[nt] = zero; }
#pragma unroll
    for (int kk = 0; kk < 4; ++kk) {
        int kb = kk * 32 + hi * 8;
        v16bf a0 = a_from_lds(h, nlo, kb);
        v16bf a1 = a_from_lds(h, 16 + nlo, kb);
        wmma_step2(c0, c1, a0, a1, Wt, 128, kk, lane);
    }
}

// Single-tile variant (global model, one wave).
__device__ __forceinline__ void next_layer1(v8f c[8], const float* bias, bool relu,
                                            const bf16t* Wt, bf16t* h, int lane) {
    int nlo = lane & 15, hi = lane >> 4;
    v8f zero = {};
#pragma unroll
    for (int nt = 0; nt < 8; ++nt) {
        int n = nt * 16 + nlo;
        float b = bias[n];
#pragma unroll
        for (int g = 0; g < 8; ++g) {
            float v = c[nt][g] + b;
            if (relu) v = fmaxf(v, 0.0f);
            h[(g + hi * 8) * 128 + n] = (bf16t)v;
        }
    }
#pragma unroll
    for (int nt = 0; nt < 8; ++nt) c[nt] = zero;
#pragma unroll
    for (int kk = 0; kk < 4; ++kk) {
        v16bf a = a_from_lds(h, nlo, kk * 32 + hi * 8);
#pragma unroll
        for (int nt = 0; nt < 8; ++nt) {
            v16bf bb = b_frag(Wt, 128, nt, kk, lane);
            c[nt] = WMMA_BF16(a, bb, c[nt]);
        }
    }
}

// ---------------- prep kernels ----------------

__global__ void zero_kernel(float* p, size_t n) {
    size_t i = (size_t)blockIdx.x * blockDim.x + threadIdx.x;
    size_t s = (size_t)gridDim.x * blockDim.x;
    for (; i < n; i += s) p[i] = 0.0f;
}

__device__ __forceinline__ void cv1(const float* s, bf16t* d, int K, int idx) {
    int n = idx / K, k = idx % K;
    d[idx] = (bf16t)s[(size_t)k * 128 + n];   // transpose [K][128] -> [128][K]
}

__global__ void __launch_bounds__(256) wcvt_kernel(
    const float* eW1, const float* eW2, const float* eW3,
    const float* nW1, const float* nW2, const float* nW3,
    const float* gW1, const float* gW2, const float* gW3, bf16t* w) {
    int id = blockIdx.x * 256 + threadIdx.x;
    if      (id <  65536) cv1(eW1, w,          512, id);
    else if (id <  81920) cv1(eW2, w +  65536, 128, id -  65536);
    else if (id <  98304) cv1(eW3, w +  81920, 128, id -  81920);
    else if (id < 147456) cv1(nW1, w +  98304, 384, id -  98304);
    else if (id < 163840) cv1(nW2, w + 147456, 128, id - 147456);
    else if (id < 180224) cv1(nW3, w + 163840, 128, id - 163840);
    else if (id < 229376) cv1(gW1, w + 180224, 384, id - 180224);
    else if (id < 245760) cv1(gW2, w + 229376, 128, id - 229376);
    else if (id < 262144) cv1(gW3, w + 245760, 128, id - 245760);
}

// ---------------- edge MLP: [E,512] -> [E,128], 32 edges/wave, scatter into agg ----------------

__global__ void __launch_bounds__(128, 1) edge_mlp_kernel(
    const float* __restrict__ x, const int* __restrict__ eidx, const float* __restrict__ ea,
    const float* __restrict__ u, const int* __restrict__ batch,
    const bf16t* W1t, const bf16t* W2t, const bf16t* W3t,
    const float* b1, const float* b2, const float* b3,
    float* __restrict__ eout, float* __restrict__ agg) {
    __shared__ __align__(32) bf16t hbuf[4][32 * 128];
    int lane = threadIdx.x & 31, wave = threadIdx.x >> 5;
    int etile = (blockIdx.x * 4 + wave) * 32;
    if (etile >= Ee) return;
    int nlo = lane & 15, hi = lane >> 4;
    int e0 = etile + nlo, e1 = etile + 16 + nlo;
    const int* rowv = eidx;
    const int* colv = eidx + Ee;
    int rs0 = rowv[e0], cs0 = colv[e0], eb0 = batch[rs0];
    int rs1 = rowv[e1], cs1 = colv[e1], eb1 = batch[rs1];

    v8f c0[8], c1[8];
    v8f zero = {};
#pragma unroll
    for (int nt = 0; nt < 8; ++nt) { c0[nt] = zero; c1[nt] = zero; }

#pragma unroll
    for (int kk = 0; kk < 16; ++kk) {                    // K = 512
        const float *s0, *s1;
        if      (kk <  4) { s0 = x  + (size_t)rs0 * 128 +  kk       * 32; s1 = x  + (size_t)rs1 * 128 +  kk       * 32; }
        else if (kk <  8) { s0 = x  + (size_t)cs0 * 128 + (kk -  4) * 32; s1 = x  + (size_t)cs1 * 128 + (kk -  4) * 32; }
        else if (kk < 12) { s0 = ea + (size_t)e0  * 128 + (kk -  8) * 32; s1 = ea + (size_t)e1  * 128 + (kk -  8) * 32; }
        else              { s0 = u  + (size_t)eb0 * 128 + (kk - 12) * 32; s1 = u  + (size_t)eb1 * 128 + (kk - 12) * 32; }
        v16bf a0 = a_from_f32(s0, lane);
        v16bf a1 = a_from_f32(s1, lane);
        wmma_step2(c0, c1, a0, a1, W1t, 512, kk, lane);
    }
    next_layer2(c0, c1, b1, true, W2t, hbuf[wave], lane);
    next_layer2(c0, c1, b2, true, W3t, hbuf[wave], lane);

#pragma unroll
    for (int t = 0; t < 2; ++t) {
        v8f* ct = t ? c1 : c0;
        int cm[8];
#pragma unroll
        for (int g = 0; g < 8; ++g) cm[g] = colv[etile + t * 16 + g + hi * 8];
#pragma unroll
        for (int nt = 0; nt < 8; ++nt) {
            int n = nt * 16 + nlo;
            float b = b3[n];
#pragma unroll
            for (int g = 0; g < 8; ++g) {
                int M = t * 16 + g + hi * 8;
                float v = ct[nt][g] + b;
                eout[(size_t)(etile + M) * 128 + n] = v;
                atomAddG(&agg[(size_t)cm[g] * 128 + n], v);
            }
        }
    }
}

// ---------------- node MLP: [N,384] -> [N,128], 32 nodes/wave ----------------

__global__ void __launch_bounds__(128, 1) node_mlp_kernel(
    const float* __restrict__ x, const float* __restrict__ agg, const float* __restrict__ u,
    const int* __restrict__ batch,
    const bf16t* W1t, const bf16t* W2t, const bf16t* W3t,
    const float* b1, const float* b2, const float* b3, float* __restrict__ xout) {
    __shared__ __align__(32) bf16t hbuf[4][32 * 128];
    int lane = threadIdx.x & 31, wave = threadIdx.x >> 5;
    int base = (blockIdx.x * 4 + wave) * 32;
    if (base >= Nn) return;
    int nlo = lane & 15, hi = lane >> 4;
    int i0 = min(base + nlo, Nn - 1);
    int i1 = min(base + 16 + nlo, Nn - 1);
    int bb0 = batch[i0], bb1 = batch[i1];

    v8f c0[8], c1[8];
    v8f zero = {};
#pragma unroll
    for (int nt = 0; nt < 8; ++nt) { c0[nt] = zero; c1[nt] = zero; }

#pragma unroll
    for (int kk = 0; kk < 12; ++kk) {                    // K = 384
        const float *s0, *s1;
        if      (kk < 4) { s0 = x   + (size_t)i0  * 128 +  kk      * 32; s1 = x   + (size_t)i1  * 128 +  kk      * 32; }
        else if (kk < 8) { s0 = agg + (size_t)i0  * 128 + (kk - 4) * 32; s1 = agg + (size_t)i1  * 128 + (kk - 4) * 32; }
        else             { s0 = u   + (size_t)bb0 * 128 + (kk - 8) * 32; s1 = u   + (size_t)bb1 * 128 + (kk - 8) * 32; }
        v16bf a0 = a_from_f32(s0, lane);
        v16bf a1 = a_from_f32(s1, lane);
        wmma_step2(c0, c1, a0, a1, W1t, 384, kk, lane);
    }
    next_layer2(c0, c1, b1, true, W2t, hbuf[wave], lane);
    next_layer2(c0, c1, b2, true, W3t, hbuf[wave], lane);

#pragma unroll
    for (int t = 0; t < 2; ++t) {
        v8f* ct = t ? c1 : c0;
#pragma unroll
        for (int nt = 0; nt < 8; ++nt) {
            int n = nt * 16 + nlo;
            float b = b3[n];
#pragma unroll
            for (int g = 0; g < 8; ++g) {
                int r = base + t * 16 + g + hi * 8;
                if (r < Nn) xout[(size_t)r * 128 + n] = ct[nt][g] + b;
            }
        }
    }
}

// ---------------- segment sums over B=8 graphs (LDS-staged) ----------------

__global__ void __launch_bounds__(256) seg_reduce_kernel(
    const float* __restrict__ vals, const int* __restrict__ rowv, const int* __restrict__ batch,
    float* __restrict__ sum, float* __restrict__ cnt, int nrows) {
    __shared__ float ls[Bb * 128];
    __shared__ float lc[Bb];
    for (int i = threadIdx.x; i < Bb * 128; i += 256) ls[i] = 0.0f;
    if (threadIdx.x < Bb) lc[threadIdx.x] = 0.0f;
    __syncthreads();
    int col = threadIdx.x & 127, rofs = threadIdx.x >> 7;     // 2 rows x 128 cols / block
    for (int r = blockIdx.x * 2 + rofs; r < nrows; r += gridDim.x * 2) {
        int b = batch[rowv ? rowv[r] : r];
        atomAddL(&ls[b * 128 + col], vals[(size_t)r * 128 + col]);
        if (col == 0) atomAddL(&lc[b], 1.0f);
    }
    __syncthreads();
    for (int i = threadIdx.x; i < Bb * 128; i += 256) atomAddG(&sum[i], ls[i]);
    if (threadIdx.x < Bb) atomAddG(&cnt[threadIdx.x], lc[threadIdx.x]);
}

// ---------------- global MLP: one wave, rows padded to 16 ----------------

__global__ void global_mlp_kernel(
    const float* __restrict__ u, const float* __restrict__ nsum, const float* __restrict__ ncnt,
    const float* __restrict__ esum, const float* __restrict__ ecnt,
    const bf16t* W1t, const bf16t* W2t, const bf16t* W3t,
    const float* b1, const float* b2, const float* b3, float* __restrict__ uout) {
    __shared__ __align__(32) float gin[16 * 384];
    __shared__ __align__(32) bf16t hb[16 * 128];
    int lane = threadIdx.x;
    for (int idx = lane; idx < 16 * 384; idx += 32) {
        int r = idx / 384, f = idx % 384;
        float v = 0.0f;
        if (r < Bb) {
            if      (f < 128) v = u[r * 128 + f];
            else if (f < 256) v = nsum[r * 128 + (f - 128)] / fmaxf(ncnt[r], 1.0f);
            else              v = esum[r * 128 + (f - 256)] / fmaxf(ecnt[r], 1.0f);
        }
        gin[idx] = v;
    }
    int nlo = lane & 15, hi = lane >> 4;
    v8f c[8];
    v8f zero = {};
#pragma unroll
    for (int nt = 0; nt < 8; ++nt) c[nt] = zero;
#pragma unroll
    for (int kk = 0; kk < 12; ++kk) {
        v16bf a = a_from_f32(gin + nlo * 384 + kk * 32, lane);
#pragma unroll
        for (int nt = 0; nt < 8; ++nt) {
            v16bf w = b_frag(W1t, 384, nt, kk, lane);
            c[nt] = WMMA_BF16(a, w, c[nt]);
        }
    }
    next_layer1(c, b1, true, W2t, hb, lane);
    next_layer1(c, b2, true, W3t, hb, lane);
    if (hi == 0) {                               // rows 0..7 live in lanes 0-15, VGPRs 0-7
#pragma unroll
        for (int nt = 0; nt < 8; ++nt) {
            int n = nt * 16 + nlo;
            float b = b3[n];
#pragma unroll
            for (int g = 0; g < 8; ++g)
                uout[(size_t)g * 128 + n] = c[nt][g] + b;
        }
    }
}

// ---------------- launch ----------------

extern "C" void kernel_launch(void* const* d_in, const int* in_sizes, int n_in,
                              void* d_out, int out_size, void* d_ws, size_t ws_size,
                              hipStream_t stream) {
    (void)in_sizes; (void)n_in; (void)out_size; (void)ws_size;
    const float* x     = (const float*)d_in[0];
    const int*   eidx  = (const int*)  d_in[1];
    const float* ea    = (const float*)d_in[2];
    const float* u     = (const float*)d_in[3];
    const int*   batch = (const int*)  d_in[4];
    const float* eW1 = (const float*)d_in[5];  const float* eb1 = (const float*)d_in[6];
    const float* eW2 = (const float*)d_in[7];  const float* eb2 = (const float*)d_in[8];
    const float* eW3 = (const float*)d_in[9];  const float* eb3 = (const float*)d_in[10];
    const float* nW1 = (const float*)d_in[11]; const float* nb1 = (const float*)d_in[12];
    const float* nW2 = (const float*)d_in[13]; const float* nb2 = (const float*)d_in[14];
    const float* nW3 = (const float*)d_in[15]; const float* nb3 = (const float*)d_in[16];
    const float* gW1 = (const float*)d_in[17]; const float* gb1 = (const float*)d_in[18];
    const float* gW2 = (const float*)d_in[19]; const float* gb2 = (const float*)d_in[20];
    const float* gW3 = (const float*)d_in[21]; const float* gb3 = (const float*)d_in[22];

    float* xout = (float*)d_out;
    float* eout = xout + (size_t)Nn * 128;
    float* uout = eout + (size_t)Ee * 128;

    float* agg  = (float*)d_ws;                 // [N,128]
    float* esum = agg  + (size_t)Nn * 128;      // [8,128]
    float* nsum = esum + Bb * 128;              // [8,128]
    float* ecnt = nsum + Bb * 128;              // [8]
    float* ncnt = ecnt + Bb;                    // [8]
    bf16t* w    = (bf16t*)(ncnt + Bb);          // byte offset 32B-aligned
    bf16t* eW1t = w;                            // 128*512
    bf16t* eW2t = eW1t + 128 * 512;
    bf16t* eW3t = eW2t + 128 * 128;
    bf16t* nW1t = eW3t + 128 * 128;             // 128*384
    bf16t* nW2t = nW1t + 128 * 384;
    bf16t* nW3t = nW2t + 128 * 128;
    bf16t* gW1t = nW3t + 128 * 128;             // 128*384
    bf16t* gW2t = gW1t + 128 * 384;
    bf16t* gW3t = gW2t + 128 * 128;

    size_t nzero = (size_t)Nn * 128 + 2 * Bb * 128 + 2 * Bb;
    zero_kernel<<<2048, 256, 0, stream>>>(agg, nzero);
    wcvt_kernel<<<1024, 256, 0, stream>>>(eW1, eW2, eW3, nW1, nW2, nW3, gW1, gW2, gW3, w);
    edge_mlp_kernel<<<Ee / 128, 128, 0, stream>>>(x, eidx, ea, u, batch,
                                                  eW1t, eW2t, eW3t, eb1, eb2, eb3, eout, agg);
    seg_reduce_kernel<<<1024, 256, 0, stream>>>(eout, eidx, batch, esum, ecnt, Ee);
    int ntiles32 = (Nn + 31) / 32;
    node_mlp_kernel<<<(ntiles32 + 3) / 4, 128, 0, stream>>>(x, agg, u, batch,
                                                            nW1t, nW2t, nW3t, nb1, nb2, nb3, xout);
    seg_reduce_kernel<<<1024, 256, 0, stream>>>(xout, (const int*)nullptr, batch, nsum, ncnt, Nn);
    global_mlp_kernel<<<1, 32, 0, stream>>>(u, nsum, ncnt, esum, ecnt,
                                            gW1t, gW2t, gW3t, gb1, gb2, gb3, uout);
}
